// SparseAttention_51049981281135
// MI455X (gfx1250) — matrix operand order repaired and checked by reading, not compiled, not run on hardware
//
#include <hip/hip_runtime.h>

// Sparse attention (B=2,H=16,S=2048,Dh=64) as masked flash-attention on gfx1250.
// Block = 4 wave32; each wave owns TWO 16-query tiles (32 queries), all 8 tiles in
// one head share double-buffered f16 K/V tiles in LDS. scores^T = K*Q^T makes the
// softmax key-reduction in-lane; K/V LDS operands are reused across both q-subtiles.
// 1/sqrt(d)*log2(e) folded into Q; softmax uses exp2 (identical after normalize).

typedef __attribute__((ext_vector_type(16))) _Float16 v16h;
typedef __attribute__((ext_vector_type(8)))  float    v8f;

#define SEQ   2048
#define DH    64
#define NWAVE 4
#define QT    2      // q-subtiles per wave

union V16 {
  v16h  h;
  uint4 q[2];
};

__device__ __forceinline__ v8f wmma16(v16h a, v16h b, v8f c) {
  // (neg_a, A, neg_b, B, c_mod, C, reuse_a, reuse_b)
  return __builtin_amdgcn_wmma_f32_16x16x32_f16(false, a, false, b, (short)0, c,
                                                false, false);
}

__device__ __forceinline__ unsigned int pkh(float a, float b) {
  union { _Float16 h[2]; unsigned int u; } t;
  t.h[0] = (_Float16)a; t.h[1] = (_Float16)b;   // fuses to v_cvt_pk_f16_f32
  return t.u;
}

// Assemble a 16-half WMMA operand from two b128 LDS loads.
__device__ __forceinline__ v16h lds2(const unsigned int* p0, const unsigned int* p1) {
  V16 t;
  t.q[0] = *(const uint4*)p0;
  t.q[1] = *(const uint4*)p1;
  return t.h;
}

// Pack 16 staged f32 into 8 u32 (f16 pairs) and store as two b128.
__device__ __forceinline__ void pack_store(unsigned int* dst, const float4* r) {
  uint4 a, b;
  a.x = pkh(r[0].x, r[0].y); a.y = pkh(r[0].z, r[0].w);
  a.z = pkh(r[1].x, r[1].y); a.w = pkh(r[1].z, r[1].w);
  b.x = pkh(r[2].x, r[2].y); b.y = pkh(r[2].z, r[2].w);
  b.z = pkh(r[3].x, r[3].y); b.w = pkh(r[3].z, r[3].w);
  ((uint4*)dst)[0] = a;
  ((uint4*)dst)[1] = b;
}

__global__ __launch_bounds__(NWAVE * 32)
void sparse_attn_kernel(const float* __restrict__ Q, const float* __restrict__ K,
                        const float* __restrict__ V,
                        const unsigned char* __restrict__ pat,
                        float* __restrict__ out) {
  __shared__ __align__(16) unsigned int kt[2][32 * 32];        // f16 [key][d]
  __shared__ __align__(16) unsigned int vt[2][32 * 32];        // f16 [key][d]
  __shared__ __align__(16) unsigned int ptile[NWAVE][QT * 256]; // f16 P^T [q][key]
  __shared__ __align__(16) float        albuf[NWAVE][QT * 16];

  const int tid  = threadIdx.x;
  const int wave = tid >> 5;
  const int lane = tid & 31;
  const int row  = lane & 15;     // M-row helper (A layout) / D-layout N index
  const int hi   = lane >> 4;
  const int col  = row;           // query-within-subtile for this lane's D columns

  const int tile0 = blockIdx.x * (NWAVE * QT) + wave * QT;  // pair of q-tiles
  const int head  = tile0 >> 7;                             // 128 q-tiles per head
  const int q0    = (tile0 & 127) << 4;                     // subtile u: q0 + 16u

  const float* Qh = Q + (size_t)head * SEQ * DH;
  const float* Kh = K + (size_t)head * SEQ * DH;
  const float* Vh = V + (size_t)head * SEQ * DH;
  const unsigned char* prow[QT] = {
      pat + (size_t)(q0 + col) * SEQ + 8 * hi,
      pat + (size_t)(q0 + 16 + col) * SEQ + 8 * hi};

  // ---- Q^T as loop-invariant B operands; fold 1/sqrt(d)*log2(e) into Q ----
  const float qscale = 0.125f * 1.4426950408889634f;
  v16h qb[QT][2];
#pragma unroll
  for (int u = 0; u < QT; ++u) {
    unsigned int u0[8], u1[8];
#pragma unroll
    for (int v = 0; v < 8; ++v) {
      const float* q2 = Qh + (size_t)(q0 + u * 16 + col) * DH + 8 * hi + 2 * v + (v >= 4 ? 8 : 0);
      u0[v] = pkh(q2[0] * qscale,  q2[1] * qscale);
      u1[v] = pkh(q2[32] * qscale, q2[33] * qscale);
    }
    V16 t0, t1;
    t0.q[0] = make_uint4(u0[0], u0[1], u0[2], u0[3]);
    t0.q[1] = make_uint4(u0[4], u0[5], u0[6], u0[7]);
    t1.q[0] = make_uint4(u1[0], u1[1], u1[2], u1[3]);
    t1.q[1] = make_uint4(u1[4], u1[5], u1[6], u1[7]);
    qb[u][0] = t0.h;
    qb[u][1] = t1.h;
  }

  // ---- cooperative staging: thread -> (key row, 16-wide d segment) ----
  const int skey = tid >> 2;          // 0..31
  const int sd   = (tid & 3) << 4;    // 0,16,32,48
  const int sidx = skey * 32 + (sd >> 1);

  float4 kreg[4], vreg[4];
  {
    const float4* kp = (const float4*)(Kh + (size_t)skey * DH + sd);
    const float4* vp = (const float4*)(Vh + (size_t)skey * DH + sd);
    kreg[0] = kp[0]; kreg[1] = kp[1]; kreg[2] = kp[2]; kreg[3] = kp[3];
    vreg[0] = vp[0]; vreg[1] = vp[1]; vreg[2] = vp[2]; vreg[3] = vp[3];
  }
  pack_store(&kt[0][sidx], kreg);
  pack_store(&vt[0][sidx], vreg);
  __syncthreads();

  v8f acc[QT][4] = {};
  float mrow[QT], lrow[QT];
#pragma unroll
  for (int u = 0; u < QT; ++u) { mrow[u] = -3.0e38f; lrow[u] = 0.0f; }
  int cur = 0;

  unsigned int* pt = ptile[wave];
  float*        ab = albuf[wave];

#pragma unroll 2
  for (int kb0 = 0; kb0 < SEQ; kb0 += 32) {
    const bool more = (kb0 + 32) < SEQ;
    if (more) {  // issue next tile's global loads early
      const float4* kp = (const float4*)(Kh + (size_t)(kb0 + 32 + skey) * DH + sd);
      const float4* vp = (const float4*)(Vh + (size_t)(kb0 + 32 + skey) * DH + sd);
      kreg[0] = kp[0]; kreg[1] = kp[1]; kreg[2] = kp[2]; kreg[3] = kp[3];
      vreg[0] = vp[0]; vreg[1] = vp[1]; vreg[2] = vp[2]; vreg[3] = vp[3];
    }

    const unsigned int* kb = kt[cur];
    const unsigned int* vb = vt[cur];

    // ---- shared K A-operands: two 16-key m-tiles x two contraction chunks ----
    v16h ka[2][2];
    ka[0][0] = lds2(kb + row * 32 + hi * 4,             kb + row * 32 + 8 + hi * 4);
    ka[0][1] = lds2(kb + row * 32 + 16 + hi * 4,        kb + row * 32 + 24 + hi * 4);
    ka[1][0] = lds2(kb + (16 + row) * 32 + hi * 4,      kb + (16 + row) * 32 + 8 + hi * 4);
    ka[1][1] = lds2(kb + (16 + row) * 32 + 16 + hi * 4, kb + (16 + row) * 32 + 24 + hi * 4);

#pragma unroll
    for (int u = 0; u < QT; ++u) {
      // ---- scores^T = K * Q^T for this q-subtile ----
      v8f st0 = {}, st1 = {};
      st0 = wmma16(ka[0][0], qb[u][0], st0);
      st0 = wmma16(ka[0][1], qb[u][1], st0);
      st1 = wmma16(ka[1][0], qb[u][0], st1);
      st1 = wmma16(ka[1][1], qb[u][1], st1);

      // ---- mask: lane holds (key = kb0 + t*16 + r + 8*hi, q) ----
      const uint2 pb0 = *(const uint2*)(prow[u] + kb0);
      const uint2 pb1 = *(const uint2*)(prow[u] + kb0 + 16);
#pragma unroll
      for (int r = 0; r < 8; ++r) {
        const unsigned w0 = (r < 4) ? pb0.x : pb0.y;
        const unsigned w1 = (r < 4) ? pb1.x : pb1.y;
        const int sh = 8 * (r & 3);
        st0[r] = ((w0 >> sh) & 0xffu) ? st0[r] : -1.0e9f;
        st1[r] = ((w1 >> sh) & 0xffu) ? st1[r] : -1.0e9f;
      }

      // ---- online softmax (log2 domain): in-lane tree + one half-merge ----
      float x0 = fmaxf(st0[0], st1[0]), x1 = fmaxf(st0[1], st1[1]);
      float x2 = fmaxf(st0[2], st1[2]), x3 = fmaxf(st0[3], st1[3]);
      float x4 = fmaxf(st0[4], st1[4]), x5 = fmaxf(st0[5], st1[5]);
      float x6 = fmaxf(st0[6], st1[6]), x7 = fmaxf(st0[7], st1[7]);
      x0 = fmaxf(x0, x1); x2 = fmaxf(x2, x3); x4 = fmaxf(x4, x5); x6 = fmaxf(x6, x7);
      float bm = fmaxf(fmaxf(x0, x2), fmaxf(x4, x6));
      bm = fmaxf(bm, __shfl_xor(bm, 16, 32));
      const float mnew  = fmaxf(mrow[u], bm);
      const float alpha = __builtin_amdgcn_exp2f(mrow[u] - mnew);
      mrow[u] = mnew;
#pragma unroll
      for (int r = 0; r < 8; ++r) {
        st0[r] = __builtin_amdgcn_exp2f(st0[r] - mnew);
        st1[r] = __builtin_amdgcn_exp2f(st1[r] - mnew);
      }
      float s0 = (st0[0] + st1[0]) + (st0[1] + st1[1]);
      float s1 = (st0[2] + st1[2]) + (st0[3] + st1[3]);
      float s2 = (st0[4] + st1[4]) + (st0[5] + st1[5]);
      float s3 = (st0[6] + st1[6]) + (st0[7] + st1[7]);
      float rs = (s0 + s1) + (s2 + s3);
      rs += __shfl_xor(rs, 16, 32);
      lrow[u] = lrow[u] * alpha + rs;

      // ---- P^T -> LDS as f16 (keys contiguous per lane: one b128 per tile) ----
      uint4 a, b;
      a.x = pkh(st0[0], st0[1]); a.y = pkh(st0[2], st0[3]);
      a.z = pkh(st0[4], st0[5]); a.w = pkh(st0[6], st0[7]);
      b.x = pkh(st1[0], st1[1]); b.y = pkh(st1[2], st1[3]);
      b.z = pkh(st1[4], st1[5]); b.w = pkh(st1[6], st1[7]);
      *(uint4*)(pt + u * 256 + col * 16 + hi * 4)     = a;
      *(uint4*)(pt + u * 256 + col * 16 + 8 + hi * 4) = b;

      if (hi == 0) ab[u * 16 + col] = alpha;
    }

    // ---- per-query alpha broadcast to D-layout rows (intra-wave via LDS) ----
#pragma unroll
    for (int u = 0; u < QT; ++u) {
      const float4 al0 = *(const float4*)(ab + u * 16 + hi * 8);
      const float4 al1 = *(const float4*)(ab + u * 16 + hi * 8 + 4);
      const float alv[8] = {al0.x, al0.y, al0.z, al0.w, al1.x, al1.y, al1.z, al1.w};
#pragma unroll
      for (int r = 0; r < 8; ++r) {
        acc[u][0][r] *= alv[r]; acc[u][1][r] *= alv[r];
        acc[u][2][r] *= alv[r]; acc[u][3][r] *= alv[r];
      }
    }

    // ---- O += P * V : V B-operands shared by both q-subtiles ----
    const v16h pa0 = lds2(pt + row * 16 + hi * 4,       pt + row * 16 + 8 + hi * 4);
    const v16h pa1 = lds2(pt + 256 + row * 16 + hi * 4, pt + 256 + row * 16 + 8 + hi * 4);
#pragma unroll
    for (int nt = 0; nt < 4; ++nt) {
      const _Float16* vh = (const _Float16*)vb + nt * 16 + row;
      v16h vop;
#pragma unroll
      for (int s = 0; s < 16; ++s)
        vop[s] = vh[((s < 8) ? (8 * hi + s) : (8 + 8 * hi + s)) * 64];
      acc[0][nt] = wmma16(pa0, vop, acc[0][nt]);
      acc[1][nt] = wmma16(pa1, vop, acc[1][nt]);
    }

    // ---- publish next K/V tile into the other buffer ----
    if (more) {
      pack_store(&kt[cur ^ 1][sidx], kreg);
      pack_store(&vt[cur ^ 1][sidx], vreg);
    }
    __syncthreads();
    cur ^= 1;
  }

  // ---- normalize + store (coalesced: n = d within tile, rows = queries) ----
#pragma unroll
  for (int u = 0; u < QT; ++u) {
    if (hi == 0) ab[u * 16 + col] = 1.0f / lrow[u];
    const float4 i0 = *(const float4*)(ab + u * 16 + hi * 8);
    const float4 i1 = *(const float4*)(ab + u * 16 + hi * 8 + 4);
    const float inv[8] = {i0.x, i0.y, i0.z, i0.w, i1.x, i1.y, i1.z, i1.w};
#pragma unroll
    for (int r = 0; r < 8; ++r) {
      float* op = out + ((size_t)head * SEQ + q0 + u * 16 + r + hi * 8) * DH + col;
      op[0]  = acc[u][0][r] * inv[r];
      op[16] = acc[u][1][r] * inv[r];
      op[32] = acc[u][2][r] * inv[r];
      op[48] = acc[u][3][r] * inv[r];
    }
  }
}

extern "C" void kernel_launch(void* const* d_in, const int* in_sizes, int n_in,
                              void* d_out, int out_size, void* d_ws, size_t ws_size,
                              hipStream_t stream) {
  (void)in_sizes; (void)n_in; (void)out_size; (void)d_ws; (void)ws_size;
  const float* Q = (const float*)d_in[0];
  const float* K = (const float*)d_in[1];
  const float* V = (const float*)d_in[2];
  const unsigned char* pat = (const unsigned char*)d_in[3];  // jax bool -> 1 byte
  float* out = (float*)d_out;

  const int ntiles = 2 * 16 * (SEQ / 16);           // B*H*(S/16) = 4096
  dim3 grid(ntiles / (NWAVE * QT));                 // 512 blocks
  dim3 block(NWAVE * 32);
  sparse_attn_kernel<<<grid, block, 0, stream>>>(Q, K, V, pat, out);
}
